// MultiHeadedAttention_25881472926108
// MI455X (gfx1250) — matrix operand order repaired and checked by reading, not compiled, not run on hardware
//
#include <hip/hip_runtime.h>

typedef __attribute__((ext_vector_type(16))) __bf16 v16bf;
typedef __attribute__((ext_vector_type(8)))  float  v8f;
typedef __attribute__((ext_vector_type(16))) short  v16s;
typedef __attribute__((ext_vector_type(8)))  short  v8s;

constexpr int B = 4, S = 2048, D = 1024, H = 16, DK = 64;
constexpr int ROWS = B * S;

__device__ __forceinline__ unsigned short f2bf(float f) {
  unsigned int x = __float_as_uint(f);
  x += 0x7FFFu + ((x >> 16) & 1u);               // round-to-nearest-even
  return (unsigned short)(x >> 16);
}

__device__ __forceinline__ v16s cat8(v8s lo, v8s hi) {
  return __builtin_shufflevector(lo, hi, 0,1,2,3,4,5,6,7,8,9,10,11,12,13,14,15);
}

__device__ __forceinline__ v8f wmma_bf16(v16s a, v16s b, v8f c) {
  return __builtin_amdgcn_wmma_f32_16x16x32_bf16(
      false, __builtin_bit_cast(v16bf, a),
      false, __builtin_bit_cast(v16bf, b),
      (short)0, c, false, false);
}

__global__ void cvt_bf16_kernel(const float* __restrict__ in,
                                unsigned short* __restrict__ out, int n) {
  int i = blockIdx.x * blockDim.x + threadIdx.x;
  if (i < n) out[i] = f2bf(in[i]);
}

// One wave computes a 32x32 tile of  out = scale * (X * W^T + bias), K = D.
// 4 WMMAs per 32-wide k-chunk with 2x reuse of both A and B fragments.
// A fragment: lane16 = row, K run 16*hi..+16 per 32-chunk.
// B fragment: lane16 = col (W row), K run 16*hi..+16 (W row-major [n][k] is
// exactly the natural B-matrix layout since B[k][n] = W[n][k]).
// MODE 0: write bf16 [B,H,S,DK]   (Q with scale=1/8, K with scale=1)
// MODE 1: write bf16 [B,H,DK,S]   (V transposed)
// MODE 2: write f32  [B,S,D]      (final projection)
template <int MODE>
__global__ __launch_bounds__(32)
void proj_gemm_kernel(const unsigned short* __restrict__ X,
                      const unsigned short* __restrict__ W,
                      const float* __restrict__ bias,
                      void* __restrict__ out, float scale) {
  const int lane   = threadIdx.x;
  const int lane16 = lane & 15;
  const int hi     = lane >> 4;
  const int m0 = blockIdx.x * 32;
  const int n0 = blockIdx.y * 32;

  const unsigned short* arow0 = X + (size_t)(m0 + lane16) * D;
  const unsigned short* arow1 = arow0 + (size_t)16 * D;
  const unsigned short* wrow0 = W + (size_t)(n0 + lane16) * D;
  const unsigned short* wrow1 = wrow0 + (size_t)16 * D;

  v8f acc[2][2] = {};
#pragma unroll 2
  for (int kb = 0; kb < D / 32; ++kb) {
    const int ka = kb * 32 + 16 * hi;
    const int kw = kb * 32 + 16 * hi;
    v16s a0 = cat8(*(const v8s*)(arow0 + ka), *(const v8s*)(arow0 + ka + 8));
    v16s a1 = cat8(*(const v8s*)(arow1 + ka), *(const v8s*)(arow1 + ka + 8));
    v16s w0 = *(const v16s*)(wrow0 + kw);
    v16s w1 = *(const v16s*)(wrow1 + kw);
    acc[0][0] = wmma_bf16(a0, w0, acc[0][0]);
    acc[0][1] = wmma_bf16(a0, w1, acc[0][1]);
    acc[1][0] = wmma_bf16(a1, w0, acc[1][0]);
    acc[1][1] = wmma_bf16(a1, w1, acc[1][1]);
  }

  const float bn0 = bias[n0 + lane16];
  const float bn1 = bias[n0 + 16 + lane16];
#pragma unroll
  for (int mt = 0; mt < 2; ++mt) {
#pragma unroll
    for (int nt = 0; nt < 2; ++nt) {
      const float bn = nt ? bn1 : bn0;
#pragma unroll
      for (int r = 0; r < 8; ++r) {
        const int m = m0 + mt * 16 + r + 8 * hi;   // C layout: M = r + 8*hi
        const int n = n0 + nt * 16 + lane16;       //            N = lane16
        const float val = (acc[mt][nt][r] + bn) * scale;
        if (MODE == 2) {
          ((float*)out)[(size_t)m * D + n] = val;
        } else {
          const int b = m >> 11, s = m & (S - 1);
          const int h = n >> 6,  dk = n & (DK - 1);
          size_t idx;
          if (MODE == 0) idx = (((size_t)(b * H + h)) * S + s) * DK + dk;
          else           idx = (((size_t)(b * H + h)) * DK + dk) * S + s;
          ((unsigned short*)out)[idx] = f2bf(val);
        }
      }
    }
  }
}

// Flash attention: one wave32 per (b, h, 16-row q tile). Streams keys in
// blocks of 64: 8 QK^T WMMAs + online softmax + 8 PV WMMAs per block.
// Scale 1/sqrt(DK) is pre-folded into Q. P is converted from the f32 C
// layout to the bf16 A layout through a 2KB LDS bounce per wave.
__global__ __launch_bounds__(32)
void attn_kernel(const unsigned short* __restrict__ Q,
                 const unsigned short* __restrict__ K,
                 const unsigned short* __restrict__ Vt,
                 const int* __restrict__ mask,
                 unsigned short* __restrict__ Xctx) {
  __shared__ unsigned short lds_p[16 * 64];

  const int lane   = threadIdx.x;
  const int lane16 = lane & 15;
  const int hi     = lane >> 4;
  const int q0 = blockIdx.x * 16;
  const int h  = blockIdx.y;
  const int b  = blockIdx.z;

  const size_t head = (size_t)(b * H + h);
  const unsigned short* Qh  = Q  + head * S * DK;
  const unsigned short* Kh  = K  + head * S * DK;
  const unsigned short* Vth = Vt + head * DK * S;
  const int* mbase = mask + (size_t)b * S * S;

  // Q A-fragments for the two 32-wide dk chunks (held for whole key loop)
  const unsigned short* qrow = Qh + (size_t)(q0 + lane16) * DK;
  v16s qa[2];
#pragma unroll
  for (int c = 0; c < 2; ++c)
    qa[c] = cat8(*(const v8s*)(qrow + c * 32 + 16 * hi),
                 *(const v8s*)(qrow + c * 32 + 8 + 16 * hi));

  v8f acc[4] = {};
  float mrow[8], lrow[8];
#pragma unroll
  for (int r = 0; r < 8; ++r) { mrow[r] = -1e30f; lrow[r] = 0.f; }

  for (int key0 = 0; key0 < S; key0 += 64) {
    // ---- scores for 4 key tiles (Q pre-scaled by 1/sqrt(DK)) ----
    v8f st[4];
#pragma unroll
    for (int t = 0; t < 4; ++t) {
      const int kb = key0 + t * 16;
      const unsigned short* krow = Kh + (size_t)(kb + lane16) * DK;
      v8f s = {};
      s = wmma_bf16(qa[0], *(const v16s*)(krow + 16 * hi), s);
      s = wmma_bf16(qa[1], *(const v16s*)(krow + 32 + 16 * hi), s);
#pragma unroll
      for (int r = 0; r < 8; ++r) {
        const int qi = q0 + r + 8 * hi;
        if (mbase[(size_t)qi * S + kb + lane16] == 0) s[r] = -1.0e9f;
      }
      st[t] = s;
    }

    // ---- online softmax (row = r + 8*hi, 16-lane-half reductions) ----
#pragma unroll
    for (int r = 0; r < 8; ++r) {
      float v = fmaxf(fmaxf(st[0][r], st[1][r]), fmaxf(st[2][r], st[3][r]));
      v = fmaxf(v, __shfl_xor(v, 1, 32));
      v = fmaxf(v, __shfl_xor(v, 2, 32));
      v = fmaxf(v, __shfl_xor(v, 4, 32));
      v = fmaxf(v, __shfl_xor(v, 8, 32));
      const float mnew  = fmaxf(mrow[r], v);
      const float alpha = __expf(mrow[r] - mnew);
      mrow[r] = mnew;
      float rs = 0.f;
#pragma unroll
      for (int t = 0; t < 4; ++t) {
        const float p = __expf(st[t][r] - mnew);
        st[t][r] = p;
        rs += p;
      }
      rs += __shfl_xor(rs, 1, 32);
      rs += __shfl_xor(rs, 2, 32);
      rs += __shfl_xor(rs, 4, 32);
      rs += __shfl_xor(rs, 8, 32);
      lrow[r] = lrow[r] * alpha + rs;
      acc[0][r] *= alpha; acc[1][r] *= alpha;
      acc[2][r] *= alpha; acc[3][r] *= alpha;
    }

    // ---- stage P (f32 C layout) -> LDS as row-major 16x64 bf16 ----
#pragma unroll
    for (int t = 0; t < 4; ++t)
#pragma unroll
      for (int r = 0; r < 8; ++r)
        lds_p[(r + 8 * hi) * 64 + t * 16 + lane16] = f2bf(st[t][r]);
    asm volatile("s_wait_dscnt 0x0" ::: "memory");

    // ---- reload P in A layout: two 32-key chunks ----
    v16s pa[2];
#pragma unroll
    for (int c = 0; c < 2; ++c)
      pa[c] = cat8(*(const v8s*)(&lds_p[lane16 * 64 + c * 32 + 16 * hi]),
                   *(const v8s*)(&lds_p[lane16 * 64 + c * 32 + 8 + 16 * hi]));

    // ---- acc += P x V (Vt rows = dk, cols = keys -> natural B frags) ----
#pragma unroll
    for (int nc = 0; nc < 4; ++nc) {
      const unsigned short* vr =
          Vth + (size_t)(nc * 16 + lane16) * S + key0 + 16 * hi;
      acc[nc] = wmma_bf16(pa[0], *(const v16s*)vr,        acc[nc]);
      acc[nc] = wmma_bf16(pa[1], *(const v16s*)(vr + 32), acc[nc]);
    }
  }

  // ---- normalize and write context as bf16 [B,S,D] ----
#pragma unroll
  for (int r = 0; r < 8; ++r) {
    const float inv = 1.0f / lrow[r];
    const int s_ = q0 + r + 8 * hi;
    unsigned short* orow = Xctx + ((size_t)b * S + s_) * D + h * DK;
#pragma unroll
    for (int nc = 0; nc < 4; ++nc)
      orow[nc * 16 + lane16] = f2bf(acc[nc][r] * inv);
  }
}

extern "C" void kernel_launch(void* const* d_in, const int* in_sizes, int n_in,
                              void* d_out, int out_size, void* d_ws, size_t ws_size,
                              hipStream_t stream) {
  (void)in_sizes; (void)n_in; (void)out_size; (void)ws_size;
  const float* query = (const float*)d_in[0];
  const float* key_  = (const float*)d_in[1];
  const float* value = (const float*)d_in[2];
  const float* Wq = (const float*)d_in[3];
  const float* bq = (const float*)d_in[4];
  const float* Wk = (const float*)d_in[5];
  const float* bk = (const float*)d_in[6];
  const float* Wv = (const float*)d_in[7];
  const float* bv = (const float*)d_in[8];
  const float* Wo = (const float*)d_in[9];
  const float* bo = (const float*)d_in[10];
  const int* mask = (const int*)d_in[11];

  const size_t NX = (size_t)B * S * D;   // 8,388,608
  const size_t NW = (size_t)D * D;       // 1,048,576
  unsigned short* Xq  = (unsigned short*)d_ws;   // bf16 copies
  unsigned short* Xk  = Xq  + NX;
  unsigned short* Xv  = Xk  + NX;
  unsigned short* Wqb = Xv  + NX;
  unsigned short* Wkb = Wqb + NW;
  unsigned short* Wvb = Wkb + NW;
  unsigned short* Wob = Wvb + NW;
  unsigned short* Qh  = Wob + NW;        // [B,H,S,DK], pre-scaled by 1/8
  unsigned short* Kh  = Qh  + NX;        // [B,H,S,DK]
  unsigned short* Vth = Kh  + NX;        // [B,H,DK,S]
  unsigned short* Xc  = Vth + NX;        // context bf16 [B,S,D]
  // total workspace: 7*16MB + 4*2MB = 120MB

  const dim3 cblk(256);
  cvt_bf16_kernel<<<dim3((unsigned)((NX + 255) / 256)), cblk, 0, stream>>>(query, Xq, (int)NX);
  cvt_bf16_kernel<<<dim3((unsigned)((NX + 255) / 256)), cblk, 0, stream>>>(key_,  Xk, (int)NX);
  cvt_bf16_kernel<<<dim3((unsigned)((NX + 255) / 256)), cblk, 0, stream>>>(value, Xv, (int)NX);
  cvt_bf16_kernel<<<dim3((unsigned)((NW + 255) / 256)), cblk, 0, stream>>>(Wq, Wqb, (int)NW);
  cvt_bf16_kernel<<<dim3((unsigned)((NW + 255) / 256)), cblk, 0, stream>>>(Wk, Wkb, (int)NW);
  cvt_bf16_kernel<<<dim3((unsigned)((NW + 255) / 256)), cblk, 0, stream>>>(Wv, Wvb, (int)NW);
  cvt_bf16_kernel<<<dim3((unsigned)((NW + 255) / 256)), cblk, 0, stream>>>(Wo, Wob, (int)NW);

  const dim3 gg(ROWS / 32, D / 32);      // 256 x 32 waves, 32x32 tiles
  proj_gemm_kernel<0><<<gg, dim3(32), 0, stream>>>(Xq, Wqb, bq, Qh, 0.125f);
  proj_gemm_kernel<0><<<gg, dim3(32), 0, stream>>>(Xk, Wkb, bk, Kh, 1.0f);
  proj_gemm_kernel<1><<<gg, dim3(32), 0, stream>>>(Xv, Wvb, bv, Vth, 1.0f);

  attn_kernel<<<dim3(S / 16, H, B), dim3(32), 0, stream>>>(Qh, Kh, Vth, mask, Xc);

  proj_gemm_kernel<2><<<gg, dim3(32), 0, stream>>>(Xc, Wob, bo, d_out, 1.0f);
}
